// Encoder_softpool_74869869904666
// MI455X (gfx1250) — matrix operand order repaired and checked by reading, not compiled, not run on hardware
//
#include <hip/hip_runtime.h>

typedef _Float16 f16;
typedef __attribute__((ext_vector_type(16))) _Float16 v16h;
typedef __attribute__((ext_vector_type(8)))  _Float16 v8h;
typedef __attribute__((ext_vector_type(8)))  float    v8f;

union U16 { v16h v; v8h p[2]; f16 h[16]; };

#define BATCH 32
#define NPTS  2048
#define NREG  16
#define DPN   256
#define TOPK  256
#define BN_EPS 1e-5f

static __device__ __forceinline__ float lrelu(float v) { return v > 0.f ? v : 0.2f * v; }

// ---------------------------------------------------------------------------
// Kernel 1: fused conv1(3->64)+BN+ReLU, conv2(64->128)+BN+ReLU. One thread/point.
// ---------------------------------------------------------------------------
__global__ __launch_bounds__(256)
void k_mlp12(const float* __restrict__ part,
             const float* __restrict__ w1, const float* __restrict__ cb1,
             const float* __restrict__ w2, const float* __restrict__ cb2,
             const float* __restrict__ g1, const float* __restrict__ be1,
             const float* __restrict__ m1, const float* __restrict__ va1,
             const float* __restrict__ g2, const float* __restrict__ be2,
             const float* __restrict__ m2, const float* __restrict__ va2,
             float* __restrict__ h2)
{
    int gid = blockIdx.x * blockDim.x + threadIdx.x;   // B*N = 65536
    int b = gid / NPTS, n = gid % NPTS;
    const float* pb = part + (size_t)b * 3 * NPTS + n;
    float p0 = pb[0], p1 = pb[NPTS], p2 = pb[2 * NPTS];
    float h1[64];
    for (int c = 0; c < 64; ++c) {
        float acc = w1[c * 3] * p0 + w1[c * 3 + 1] * p1 + w1[c * 3 + 2] * p2 + cb1[c];
        float s = g1[c] * rsqrtf(va1[c] + BN_EPS);
        acc = acc * s + (be1[c] - m1[c] * s);
        h1[c] = acc > 0.f ? acc : 0.f;
    }
    float* o = h2 + (size_t)b * 128 * NPTS + n;
    for (int c = 0; c < 128; ++c) {
        float acc = cb2[c];
        const float* wr = w2 + c * 64;
        #pragma unroll 8
        for (int k = 0; k < 64; ++k) acc += wr[k] * h1[k];
        float s = g2[c] * rsqrtf(va2[c] + BN_EPS);
        acc = acc * s + (be2[c] - m2[c] * s);
        o[(size_t)c * NPTS] = acc > 0.f ? acc : 0.f;
    }
}

// ---------------------------------------------------------------------------
// Kernel 2: conv3 (128->256) + BN3. LDS-staged [128 x 32] point tile.
// ---------------------------------------------------------------------------
#define TILEB 32
__global__ __launch_bounds__(256)
void k_conv3(const float* __restrict__ h2,
             const float* __restrict__ w3, const float* __restrict__ cb3,
             const float* __restrict__ g3, const float* __restrict__ be3,
             const float* __restrict__ m3, const float* __restrict__ va3,
             float* __restrict__ x)
{
    __shared__ float tile[128 * TILEB];
    int pn0 = blockIdx.x * TILEB;
    int b = pn0 / NPTS, n0 = pn0 % NPTS;
    int t = threadIdx.x;
    for (int e = t; e < 128 * TILEB; e += 256) {
        int c = e / TILEB, i = e % TILEB;
        tile[e] = h2[(size_t)(b * 128 + c) * NPTS + n0 + i];
    }
    __syncthreads();
    int row = t;  // 256 rows
    float acc[TILEB];
    #pragma unroll
    for (int i = 0; i < TILEB; ++i) acc[i] = cb3[row];
    const float* wr = w3 + row * 128;
    for (int k = 0; k < 128; ++k) {
        float wv = wr[k];
        const float* tr = tile + k * TILEB;
        #pragma unroll
        for (int i = 0; i < TILEB; ++i) acc[i] += wv * tr[i];
    }
    float s = g3[row] * rsqrtf(va3[row] + BN_EPS);
    float tt = be3[row] - m3[row] * s;
    float* o = x + (size_t)(b * DPN + row) * NPTS + n0;
    #pragma unroll
    for (int i = 0; i < TILEB; ++i) o[i] = acc[i] * s + tt;
}

// ---------------------------------------------------------------------------
// Kernel 3: sort head 256->16 + per-point argmax (strict >, jnp.argmax ties).
// ---------------------------------------------------------------------------
__global__ __launch_bounds__(256)
void k_sort(const float* __restrict__ x,
            const float* __restrict__ sw, const float* __restrict__ sb,
            float* __restrict__ val, int* __restrict__ idx_sort)
{
    int gid = blockIdx.x * blockDim.x + threadIdx.x;
    int b = gid / NPTS, n = gid % NPTS;
    float acc[NREG];
    #pragma unroll
    for (int r = 0; r < NREG; ++r) acc[r] = sb[r];
    const float* xc = x + (size_t)b * DPN * NPTS + n;
    for (int k = 0; k < DPN; ++k) {
        float xv = xc[(size_t)k * NPTS];
        #pragma unroll
        for (int r = 0; r < NREG; ++r) acc[r] += sw[r * DPN + k] * xv;
    }
    int best = 0; float bv = acc[0];
    #pragma unroll
    for (int r = 0; r < NREG; ++r) {
        val[(size_t)(b * NREG + r) * NPTS + n] = acc[r];
        if (acc[r] > bv) { bv = acc[r]; best = r; }
    }
    idx_sort[b * NPTS + n] = best;
}

// ---------------------------------------------------------------------------
// Kernel 4: sorted top-256 of 2048 per (b,region): full bitonic sort in LDS of
// 64-bit keys {~sortable(float), idx}; ascending == descending value, low-idx
// tie-break (== lax.top_k ordering).
// ---------------------------------------------------------------------------
__global__ __launch_bounds__(256)
void k_topk(const float* __restrict__ val, int* __restrict__ idx_top)
{
    __shared__ unsigned long long keys[NPTS];
    int br = blockIdx.x;           // b*16 + r
    int t = threadIdx.x;
    const float* v = val + (size_t)br * NPTS;
    for (int e = t; e < NPTS; e += 256) {
        unsigned u = __float_as_uint(v[e]);
        unsigned s = (u >> 31) ? ~u : (u | 0x80000000u);
        keys[e] = ((unsigned long long)(~s) << 32) | (unsigned)e;
    }
    __syncthreads();
    for (int size = 2; size <= NPTS; size <<= 1) {
        for (int stride = size >> 1; stride > 0; stride >>= 1) {
            #pragma unroll
            for (int q = 0; q < NPTS / 2 / 256; ++q) {
                int p = t + q * 256;
                int low = p & (stride - 1);
                int i = ((p - low) << 1) + low;
                int j = i + stride;
                unsigned long long a = keys[i], bk = keys[j];
                bool dir = ((i & size) == 0);
                if ((a > bk) == dir) { keys[i] = bk; keys[j] = a; }
            }
            __syncthreads();
        }
    }
    idx_top[br * TOPK + t] = (int)(unsigned)(keys[t] & 0xffffffffu);
}

// ---------------------------------------------------------------------------
// Kernel 5: gathers -> input_chosen (output 1) + f16 feat [B,272,4096].
// ---------------------------------------------------------------------------
__global__ __launch_bounds__(256)
void k_gather(const float* __restrict__ part, const float* __restrict__ x,
              const int* __restrict__ idx_top, const int* __restrict__ idx_sort,
              float* __restrict__ out1, f16* __restrict__ feat)
{
    int gid = blockIdx.x * blockDim.x + threadIdx.x;  // B*4096
    int b = gid >> 12, j = gid & 4095;
    int idx = idx_top[(size_t)b * 4096 + j];
    #pragma unroll
    for (int d = 0; d < 3; ++d)
        out1[((size_t)b * 4096 + j) * 3 + d] = part[(size_t)(b * 3 + d) * NPTS + idx];
    const float* xc = x + (size_t)b * DPN * NPTS + idx;
    f16* fc = feat + (size_t)b * (DPN + NREG) * 4096 + j;
    for (int c = 0; c < DPN; ++c)
        fc[(size_t)c * 4096] = (f16)xc[(size_t)c * NPTS];
    int r2 = idx_sort[b * NPTS + idx];
    #pragma unroll
    for (int rr = 0; rr < NREG; ++rr)
        fc[(size_t)(DPN + rr) * 4096] = (f16)(rr == r2 ? 1.f : 0.f);
}

// ---------------------------------------------------------------------------
// Kernel 6a: pre-pack conv weights -> f16 row-major [O][Kp], K zero-padded to
// a multiple of 32 so A-fragments become two contiguous b128 loads per lane.
// ---------------------------------------------------------------------------
__global__ __launch_bounds__(256)
void k_packw(const float* __restrict__ W, f16* __restrict__ out, int Ktot, int Kp)
{
    int gid = blockIdx.x * blockDim.x + threadIdx.x;   // O*Kp
    int m = gid / Kp, k = gid - m * Kp;
    out[gid] = (k < Ktot) ? (f16)W[(size_t)m * Ktot + k] : (f16)0.f;
}

// ---------------------------------------------------------------------------
// Kernel 6b: K-major im2col (k = 3c+t, replicate pad, stride 2), zero-padded
// to Kp. One column (output position) per thread; writes are contiguous and
// B-fragments become one aligned 32B v16h load per lane.
// ---------------------------------------------------------------------------
__global__ __launch_bounds__(256)
void k_im2col(const f16* __restrict__ in, f16* __restrict__ Bp,
              int C, int Win, int Kp)
{
    int Nout = Win >> 1;
    int gid = blockIdx.x * blockDim.x + threadIdx.x;   // B*Nout
    int b = gid / Nout, n = gid - b * Nout;
    const f16* ib = in + (size_t)b * C * Win;
    f16* oc = Bp + ((size_t)b * Nout + n) * Kp;
    int c0 = 2 * n - 1;
    int colm = c0 < 0 ? 0 : c0;
    int colp = (c0 + 2) >= Win ? Win - 1 : c0 + 2;
    for (int c = 0; c < C; ++c) {
        const f16* row = ib + (size_t)c * Win;
        oc[3 * c]     = row[colm];
        oc[3 * c + 1] = row[c0 + 1];
        oc[3 * c + 2] = row[colp];
    }
    for (int k = 3 * C; k < Kp; ++k) oc[k] = (f16)0.f;
}

// ---------------------------------------------------------------------------
// Kernel 6c: WMMA GEMM on packed f16: out[b,m,n] = lrelu(A[m,:]·B[b,n,:]+bias).
// Wave computes a 16(M) x 64(N) block: per 32-wide K chunk -> 2 A b128 loads,
// 4 B (2xb128) loads, 4 x v_wmma_f32_16x16x32_f16.
// ---------------------------------------------------------------------------
__global__ __launch_bounds__(32)
void k_gemm_wmma(const f16* __restrict__ Ap, const float* __restrict__ bias,
                 const f16* __restrict__ Bp, f16* __restrict__ out,
                 int O, int Kp, int Nout)
{
    int lane = threadIdx.x, ml = lane & 15, hi = lane >> 4;
    int b = blockIdx.z;
    int mb = blockIdx.y * 16;
    int n0 = blockIdx.x * 64;
    const f16* Arow = Ap + (size_t)(mb + ml) * Kp + 16 * hi;
    const f16* B0 = Bp + ((size_t)b * Nout + n0 + ml) * Kp + 16 * hi;
    const f16* B1 = B0 + (size_t)16 * Kp;
    const f16* B2 = B0 + (size_t)32 * Kp;
    const f16* B3 = B0 + (size_t)48 * Kp;
    v8f acc0 = {}, acc1 = {}, acc2 = {}, acc3 = {};
    for (int kk = 0; kk < Kp; kk += 32) {
        U16 A;
        A.p[0] = *(const v8h*)(Arow + kk);
        A.p[1] = *(const v8h*)(Arow + kk + 8);
        v16h b0 = *(const v16h*)(B0 + kk);
        v16h b1 = *(const v16h*)(B1 + kk);
        v16h b2 = *(const v16h*)(B2 + kk);
        v16h b3 = *(const v16h*)(B3 + kk);
        acc0 = __builtin_amdgcn_wmma_f32_16x16x32_f16(false, A.v, false, b0, (short)0, acc0, false, false);
        acc1 = __builtin_amdgcn_wmma_f32_16x16x32_f16(false, A.v, false, b1, (short)0, acc1, false, false);
        acc2 = __builtin_amdgcn_wmma_f32_16x16x32_f16(false, A.v, false, b2, (short)0, acc2, false, false);
        acc3 = __builtin_amdgcn_wmma_f32_16x16x32_f16(false, A.v, false, b3, (short)0, acc3, false, false);
    }
    #pragma unroll
    for (int vv = 0; vv < 8; ++vv) {
        int m = mb + vv + 8 * hi;
        float bi = bias[m];
        f16* ob = out + ((size_t)b * O + m) * Nout + n0 + ml;
        ob[0]  = (f16)lrelu(acc0[vv] + bi);
        ob[16] = (f16)lrelu(acc1[vv] + bi);
        ob[32] = (f16)lrelu(acc2[vv] + bi);
        ob[48] = (f16)lrelu(acc3[vv] + bi);
    }
}

// ---------------------------------------------------------------------------
// Kernel 7: both maxpools fused (contiguous 256-wide reduce) -> p f16 [64,512].
// ---------------------------------------------------------------------------
__global__ __launch_bounds__(256)
void k_pool(const f16* __restrict__ y3, f16* __restrict__ p)
{
    int gid = blockIdx.x * blockDim.x + threadIdx.x;  // B*512*2 = 32768
    int w = gid & 1, i = (gid >> 1) & 511, b = gid >> 10;
    const f16* c = y3 + (size_t)(b * 512 + i) * 512 + w * 256;
    float mv = -3.4e38f;
    for (int e = 0; e < 256; ++e) { float vv = (float)c[e]; mv = vv > mv ? vv : mv; }
    p[(size_t)(b * 2 + w) * 512 + i] = (f16)mv;
}

// ---------------------------------------------------------------------------
// Kernel 8: ConvTranspose as GEMM [64 x 512] x [512 x 262144] via WMMA.
// Single streaming pass over the 536MB dec_w (nontemporal, every 64B segment
// fully consumed); B-fragment shared by all 4 M-tiles; A = two b128 loads.
// ---------------------------------------------------------------------------
__global__ __launch_bounds__(32)
void k_dec_wmma(const f16* __restrict__ p, const float* __restrict__ Wd,
                const float* __restrict__ bd, float* __restrict__ out)
{
    int lane = threadIdx.x;
    int ml = lane & 15, hi = lane >> 4;
    int n = (blockIdx.x << 4) + ml;       // 0 .. 262143
    int bbase = hi ? 16 : 0;
    const f16* pr0 = p + (size_t)(ml) * 512 + 8 * hi;
    const f16* pr1 = pr0 + 16 * 512;
    const f16* pr2 = pr0 + 32 * 512;
    const f16* pr3 = pr0 + 48 * 512;
    v8f acc0 = {}, acc1 = {}, acc2 = {}, acc3 = {};
    for (int kk = 0; kk < 512; kk += 32) {
        if (kk + 32 < 512)
            __builtin_prefetch(&Wd[(size_t)(kk + 32 + bbase) * 262144 + n], 0, 1);
        U16 Bf, A0, A1, A2, A3;
        #pragma unroll
        for (int jj = 0; jj < 16; ++jj) {
            int i = kk + (jj < 8 ? 8 * hi + jj : 16 + 8 * hi + (jj - 8));
            Bf.h[jj] = (f16)__builtin_nontemporal_load(&Wd[(size_t)i * 262144 + n]);
        }
        A0.p[0] = *(const v8h*)(pr0 + kk); A0.p[1] = *(const v8h*)(pr0 + kk + 16);
        A1.p[0] = *(const v8h*)(pr1 + kk); A1.p[1] = *(const v8h*)(pr1 + kk + 16);
        A2.p[0] = *(const v8h*)(pr2 + kk); A2.p[1] = *(const v8h*)(pr2 + kk + 16);
        A3.p[0] = *(const v8h*)(pr3 + kk); A3.p[1] = *(const v8h*)(pr3 + kk + 16);
        acc0 = __builtin_amdgcn_wmma_f32_16x16x32_f16(false, A0.v, false, Bf.v, (short)0, acc0, false, false);
        acc1 = __builtin_amdgcn_wmma_f32_16x16x32_f16(false, A1.v, false, Bf.v, (short)0, acc1, false, false);
        acc2 = __builtin_amdgcn_wmma_f32_16x16x32_f16(false, A2.v, false, Bf.v, (short)0, acc2, false, false);
        acc3 = __builtin_amdgcn_wmma_f32_16x16x32_f16(false, A3.v, false, Bf.v, (short)0, acc3, false, false);
    }
    int o = n >> 9, kcol = n & 511;
    float bo = bd[o];
    #pragma unroll
    for (int vv = 0; vv < 8; ++vv) {
        int mv = vv + (hi ? 8 : 0);
        int m;
        m = mv;      out[((size_t)(m >> 1) * 512 + o) * 1024 + (m & 1) * 512 + kcol] = lrelu(acc0[vv] + bo);
        m = 16 + mv; out[((size_t)(m >> 1) * 512 + o) * 1024 + (m & 1) * 512 + kcol] = lrelu(acc1[vv] + bo);
        m = 32 + mv; out[((size_t)(m >> 1) * 512 + o) * 1024 + (m & 1) * 512 + kcol] = lrelu(acc2[vv] + bo);
        m = 48 + mv; out[((size_t)(m >> 1) * 512 + o) * 1024 + (m & 1) * 512 + kcol] = lrelu(acc3[vv] + bo);
    }
}

// ---------------------------------------------------------------------------
extern "C" void kernel_launch(void* const* d_in, const int* in_sizes, int n_in,
                              void* d_out, int out_size, void* d_ws, size_t ws_size,
                              hipStream_t stream)
{
    (void)in_sizes; (void)n_in; (void)out_size; (void)ws_size;
    const float* part = (const float*)d_in[0];
    const float* c1w = (const float*)d_in[1];  const float* c1b = (const float*)d_in[2];
    const float* c2w = (const float*)d_in[3];  const float* c2b = (const float*)d_in[4];
    const float* c3w = (const float*)d_in[5];  const float* c3b = (const float*)d_in[6];
    const float* g1  = (const float*)d_in[7];  const float* be1 = (const float*)d_in[8];
    const float* m1  = (const float*)d_in[9];  const float* va1 = (const float*)d_in[10];
    const float* g2  = (const float*)d_in[11]; const float* be2 = (const float*)d_in[12];
    const float* m2  = (const float*)d_in[13]; const float* va2 = (const float*)d_in[14];
    const float* g3  = (const float*)d_in[15]; const float* be3 = (const float*)d_in[16];
    const float* m3  = (const float*)d_in[17]; const float* va3 = (const float*)d_in[18];
    const float* sw  = (const float*)d_in[19]; const float* sb  = (const float*)d_in[20];
    const float* e1w = (const float*)d_in[21]; const float* e1b = (const float*)d_in[22];
    const float* e2w = (const float*)d_in[23]; const float* e2b = (const float*)d_in[24];
    const float* e3w = (const float*)d_in[25]; const float* e3b = (const float*)d_in[26];
    const float* dw  = (const float*)d_in[27]; const float* db  = (const float*)d_in[28];

    char* ws = (char*)d_ws;
    // ---- workspace layout (bytes) ----
    float* x    = (float*)(ws + 0);              //  64.00 MiB [B,256,N] f32
    float* val  = (float*)(ws + 67108864ULL);    //   4.00 MiB [B,16,N]  f32
    float* h2   = (float*)(ws + 71303168ULL);    //  32.00 MiB [B,128,N] f32
    f16*   feat = (f16*)  (ws + 104857600ULL);   //  68.00 MiB [B,272,4096] f16
    int*   idxs = (int*)  (ws + 176160768ULL);   //  [B,N]
    int*   idxt = (int*)  (ws + 176422912ULL);   //  [B,16,256]
    f16*   Bp   = (f16*)  (ws + 176947200ULL);   // 104.00 MiB im2col scratch (max layer)
    f16*   A1   = (f16*)  (ws + 285999104ULL);   //  [256][832]  f16
    f16*   A2   = (f16*)  (ws + 286425088ULL);   //  [512][768]  f16
    f16*   A3   = (f16*)  (ws + 287211520ULL);   //  [512][1536] f16
    // aliases of dead buffers:
    f16*   y1   = (f16*)h2;                      //  [B,256,2048] f16
    f16*   y2   = (f16*)x;                       //  [B,512,1024] f16
    f16*   y3   = (f16*)(ws + 33554432ULL);      //  [B,512,512]  f16
    f16*   p    = (f16*)val;                     //  [64,512] f16

    float* out1 = (float*)d_out;                        // input_chosen [B,4096,3]
    float* out2 = out1 + (size_t)BATCH * 4096 * 3;      // feat_out [B,512,1024]

    // point MLP + softpool front-end
    k_mlp12<<<BATCH * NPTS / 256, 256, 0, stream>>>(part, c1w, c1b, c2w, c2b,
                                                    g1, be1, m1, va1, g2, be2, m2, va2, h2);
    k_conv3<<<BATCH * NPTS / TILEB, 256, 0, stream>>>(h2, c3w, c3b, g3, be3, m3, va3, x);
    k_sort<<<BATCH * NPTS / 256, 256, 0, stream>>>(x, sw, sb, val, idxs);
    k_topk<<<BATCH * NREG, 256, 0, stream>>>(val, idxt);
    k_gather<<<BATCH * 4096 / 256, 256, 0, stream>>>(part, x, idxt, idxs, out1, feat);

    // weight pre-pack (tiny)
    k_packw<<<256 * 832 / 256, 256, 0, stream>>>(e1w, A1, 816, 832);
    k_packw<<<512 * 768 / 256, 256, 0, stream>>>(e2w, A2, 768, 768);
    k_packw<<<512 * 1536 / 256, 256, 0, stream>>>(e3w, A3, 1536, 1536);

    // enc1: [B,272,4096] -> [B,256,2048]
    k_im2col<<<BATCH * 2048 / 256, 256, 0, stream>>>(feat, Bp, 272, 4096, 832);
    k_gemm_wmma<<<dim3(2048 / 64, 256 / 16, BATCH), 32, 0, stream>>>(A1, e1b, Bp, y1, 256, 832, 2048);
    // enc2: [B,256,2048] -> [B,512,1024]
    k_im2col<<<BATCH * 1024 / 256, 256, 0, stream>>>(y1, Bp, 256, 2048, 768);
    k_gemm_wmma<<<dim3(1024 / 64, 512 / 16, BATCH), 32, 0, stream>>>(A2, e2b, Bp, y2, 512, 768, 1024);
    // enc3: [B,512,1024] -> [B,512,512]
    k_im2col<<<BATCH * 512 / 256, 256, 0, stream>>>(y2, Bp, 512, 1024, 1536);
    k_gemm_wmma<<<dim3(512 / 64, 512 / 16, BATCH), 32, 0, stream>>>(A3, e3b, Bp, y3, 512, 1536, 512);

    // pooling + transposed-conv GEMM
    k_pool<<<BATCH * 512 * 2 / 256, 256, 0, stream>>>(y3, p);
    k_dec_wmma<<<262144 / 16, 32, 0, stream>>>(p, dw, db, out2);
}
